// PVMLayer8_6511170421520
// MI455X (gfx1250) — compile-verified, matrix operand order checked by reading, not run on hardware
//
#include <hip/hip_runtime.h>
#include <hip/hip_bf16.h>

typedef __attribute__((ext_vector_type(16))) _Float16 v16h;
typedef __attribute__((ext_vector_type(8)))  _Float16 h8;
typedef __attribute__((ext_vector_type(8)))  float    v8f;

#define LL 4096
#define NSEQ 32
#define ROWS (NSEQ * LL)   // 131072 tokens
#define CHN 16             // scan chunks
#define CLEN (LL / CHN)    // 256

// ---------------- workspace layout (bytes) ----------------
static constexpr size_t OFF_XM_F32 = 0;                         // 4M f32
static constexpr size_t OFF_XM_H   = OFF_XM_F32 + 16777216;     // 4M f16
static constexpr size_t OFF_XZ     = OFF_XM_H   + 8388608;      // 16M f32 (xc|z)
static constexpr size_t OFF_XCS_H  = OFF_XZ     + 67108864;     // 8M f16
static constexpr size_t OFF_XDBL   = OFF_XCS_H  + 16777216;     // 131072*48 f32 (B|C|dt)
static constexpr size_t OFF_DT     = OFF_XDBL   + 25165824;     // 8M f32
static constexpr size_t OFF_YS     = OFF_DT     + 33554432;     // 8M f32
static constexpr size_t OFF_YG_H   = OFF_YS     + 33554432;     // 8M f16
static constexpr size_t OFF_YM     = OFF_YG_H   + 16777216;     // 4M f32
static constexpr size_t OFF_YN_H   = OFF_YM     + 16777216;     // 4M f16
static constexpr size_t OFF_EB     = OFF_YN_H   + 8388608;      // 512K f32
static constexpr size_t OFF_PB     = OFF_EB     + 2097152;
static constexpr size_t OFF_H0     = OFF_PB     + 2097152;
static constexpr size_t OFF_WIN    = OFF_H0     + 2097152;      // 128x32 f16
static constexpr size_t OFF_WXP    = OFF_WIN    + 8192;         // 48x64 f16 (padded)
static constexpr size_t OFF_WOUT   = OFF_WXP    + 6144;         // 32x64 f16
static constexpr size_t OFF_WPJ    = OFF_WOUT   + 4096;         // 256x256 f16
static constexpr size_t OFF_ANEG   = OFF_WPJ    + 131072;       // 64x16 f32

// ---------------- WMMA fragment helpers (CDNA5 16-bit layout) ----------------
// 16-bit A 16x32 / B 32x16 fragment: lane idx=lane&15 picks row(A)/col(B),
// g=lane>>4 picks K-half; halves 0..7 = K at kbase+8g, halves 8..15 = kbase+16+8g.
__device__ __forceinline__ v16h ldfrag(const _Float16* base, int rowcol, int ld, int kbase) {
    int lane = threadIdx.x & 31;
    int g = lane >> 4, idx = lane & 15;
    const _Float16* p = base + (size_t)(rowcol + idx) * ld + kbase + 8 * g;
    h8 lo = *(const h8*)p;
    h8 hi = *(const h8*)(p + 16);
    return __builtin_shufflevector(lo, hi, 0,1,2,3,4,5,6,7,8,9,10,11,12,13,14,15);
}

__device__ __forceinline__ v8f wmma32(v16h a, v16h b, v8f c) {
    return __builtin_amdgcn_wmma_f32_16x16x32_f16(false, a, false, b, (short)0, c, false, false);
}

// ---------------- block-wide LayerNorm stats (wave32) ----------------
__device__ __forceinline__ void block_stats(float v, float& mean, float& rstd, float* sm) {
    float s1 = v, s2 = v * v;
    #pragma unroll
    for (int o = 16; o; o >>= 1) { s1 += __shfl_xor(s1, o, 32); s2 += __shfl_xor(s2, o, 32); }
    int wid = threadIdx.x >> 5, lane = threadIdx.x & 31;
    if (lane == 0) { sm[wid] = s1; sm[8 + wid] = s2; }
    __syncthreads();
    if (wid == 0) {
        float a = (lane < 8) ? sm[lane] : 0.f;
        float b = (lane < 8) ? sm[8 + lane] : 0.f;
        #pragma unroll
        for (int o = 4; o; o >>= 1) { a += __shfl_xor(a, o, 32); b += __shfl_xor(b, o, 32); }
        if (lane == 0) { sm[16] = a; sm[17] = b; }
    }
    __syncthreads();
    float mu = sm[16] * (1.f / 256.f);
    float var = sm[17] * (1.f / 256.f) - mu * mu;
    mean = mu; rstd = rsqrtf(var + 1e-5f);
}

// ---------------- K0: weight prep (f16 packs, A = -exp(A_log)) ----------------
__global__ void k_prep(const float* inw, const float* xpw, const float* outw,
                       const float* pw, const float* alog,
                       _Float16* Win, _Float16* Wxp, _Float16* Wout, _Float16* Wpj, float* Aneg) {
    int i = blockIdx.x * 256 + threadIdx.x;
    if (i < 4096) { Win[i] = (_Float16)inw[i]; }
    else if (i < 7168) { int j = i - 4096; int r = j >> 6, k = j & 63;
        Wxp[j] = (_Float16)(r < 34 ? xpw[r * 64 + k] : 0.f); }
    else if (i < 9216) { int j = i - 7168; Wout[j] = (_Float16)outw[j]; }
    else if (i < 74752) { int j = i - 9216; Wpj[j] = (_Float16)pw[j]; }
    else if (i < 75776) { int j = i - 74752; Aneg[j] = -__expf(alog[j]); }
}

// ---------------- K1: LayerNorm over C=256, rechunk to (32, L, 32) ----------------
__global__ void k_ln1(const float* __restrict__ x, const float* nw, const float* nb,
                      float* xmf, _Float16* xmh) {
    __shared__ float sm[18];
    int blk = blockIdx.x; int b = blk >> 12; int l = blk & 4095; int c = threadIdx.x;
    float v = x[(size_t)(b * 256 + c) * 4096 + l];
    float mu, rs; block_stats(v, mu, rs, sm);
    float xm = (v - mu) * rs * nw[c] + nb[c];
    size_t o = ((size_t)(b * 8 + (c >> 5)) * 4096 + l) * 32 + (c & 31);
    xmf[o] = xm; xmh[o] = (_Float16)xm;
}

// ---------------- K2: in_proj GEMM (131072x32)x(32->128); A held in regs, 8 WMMA/wave ----
__global__ void k_inproj(const _Float16* __restrict__ A, const _Float16* __restrict__ W,
                         float* __restrict__ XZ) {
    int w = threadIdx.x >> 5, lane = threadIdx.x & 31;
    int mrow = (blockIdx.x * 8 + w) * 16;
    v16h a = ldfrag(A, mrow, 32, 0);
    int mb = mrow + ((lane >> 4) << 3);
    int cl = lane & 15;
    #pragma unroll
    for (int nt = 0; nt < 8; nt++) {
        v16h b = ldfrag(W, nt * 16, 32, 0);
        v8f c = {};
        c = wmma32(a, b, c);
        int col = nt * 16 + cl;
        #pragma unroll
        for (int r = 0; r < 8; r++) XZ[(size_t)(mb + r) * 128 + col] = c[r];
    }
}

// ---------------- K3: causal depthwise conv(4) + SiLU ----------------
__global__ void k_conv(const float* __restrict__ XZ, const float* cw, const float* cb,
                       _Float16* __restrict__ XCS) {
    size_t i = (size_t)blockIdx.x * 256 + threadIdx.x;   // < 131072*64
    int d = (int)(i & 63); size_t t = i >> 6;
    int l = (int)(t & 4095); size_t nb0 = t - l;
    float acc = cb[d];
    #pragma unroll
    for (int k = 0; k < 4; k++) {
        int ll = l - 3 + k;
        if (ll >= 0) acc += cw[d * 4 + k] * XZ[(nb0 + ll) * 128 + d];
    }
    float s = acc * (1.f / (1.f + __expf(-acc)));
    XCS[i] = (_Float16)s;
}

// ---------------- K4: x_proj GEMM (64 -> 48); 2 K-frags in regs, 6 WMMA/wave ----------
__global__ void k_xproj(const _Float16* __restrict__ A, const _Float16* __restrict__ W,
                        float* __restrict__ XDBL) {
    int w = threadIdx.x >> 5, lane = threadIdx.x & 31;
    int mrow = (blockIdx.x * 8 + w) * 16;
    v16h a0 = ldfrag(A, mrow, 64, 0);
    v16h a1 = ldfrag(A, mrow, 64, 32);
    int mb = mrow + ((lane >> 4) << 3);
    int cl = lane & 15;
    #pragma unroll
    for (int nt = 0; nt < 3; nt++) {
        v8f c = {};
        c = wmma32(a0, ldfrag(W, nt * 16, 64, 0), c);
        c = wmma32(a1, ldfrag(W, nt * 16, 64, 32), c);
        int col = nt * 16 + cl;
        int mapped = (col < 2) ? 32 + col : (col < 34 ? col - 2 : -1);
        if (mapped >= 0) {
            #pragma unroll
            for (int r = 0; r < 8; r++) XDBL[(size_t)(mb + r) * 48 + mapped] = c[r];
        }
    }
}

// ---------------- K5: dt = softplus(dt2 @ dt_proj_w^T + b) ----------------
__global__ void k_dt(const float* __restrict__ XDBL, const float* dtw, const float* dtb,
                     float* __restrict__ DT) {
    size_t i = (size_t)blockIdx.x * 256 + threadIdx.x;
    int d = (int)(i & 63); size_t row = i >> 6;
    const float* xr = XDBL + row * 48;
    float v = xr[32] * dtw[d * 2] + xr[33] * dtw[d * 2 + 1] + dtb[d];
    DT[i] = (v > 20.f) ? v : log1pf(__expf(v));
}

// ---------------- selective scan: 3-phase chunked parallel scan ----------------
union F16U { float4 q[4]; float f[16]; };

__global__ void k_scan1(const float* __restrict__ DT, const float* __restrict__ XDBL,
                        const _Float16* __restrict__ XCS, const float* __restrict__ Aneg,
                        float* __restrict__ Eb, float* __restrict__ Pb) {
    int unit = blockIdx.x * 4 + (threadIdx.x >> 6);       // n*CHN + ch
    int n = unit >> 4, ch = unit & 15, d = threadIdx.x & 63;
    float Ad[16];
    #pragma unroll
    for (int s = 0; s < 16; s++) Ad[s] = Aneg[d * 16 + s];
    float h[16], P[16];
    #pragma unroll
    for (int s = 0; s < 16; s++) { h[s] = 0.f; P[s] = 1.f; }
    size_t row0 = (size_t)n * LL + ch * CLEN;
    for (int l = 0; l < CLEN; l++) {
        size_t row = row0 + l;
        size_t prow = row + 8 < (size_t)ROWS ? row + 8 : row;
        __builtin_prefetch(XDBL + prow * 48, 0, 1);
        __builtin_prefetch(DT + prow * 64 + d, 0, 1);
        float dt = DT[row * 64 + d];
        float dx = dt * (float)XCS[row * 64 + d];
        F16U U; const float4* pq = (const float4*)(XDBL + row * 48);
        U.q[0] = pq[0]; U.q[1] = pq[1]; U.q[2] = pq[2]; U.q[3] = pq[3];
        #pragma unroll
        for (int s = 0; s < 16; s++) {
            float a = __expf(dt * Ad[s]);
            h[s] = h[s] * a + dx * U.f[s];
            P[s] *= a;
        }
    }
    size_t base = ((size_t)unit * 64 + d) * 16;
    #pragma unroll
    for (int s = 0; s < 16; s++) { Eb[base + s] = h[s]; Pb[base + s] = P[s]; }
}

__global__ void k_scan2(const float* __restrict__ Eb, const float* __restrict__ Pb,
                        float* __restrict__ H0) {
    int i = blockIdx.x * 256 + threadIdx.x;
    if (i >= NSEQ * 64 * 16) return;
    int s = i & 15, d = (i >> 4) & 63, n = i >> 10;
    float H = 0.f;
    for (int c = 0; c < CHN; c++) {
        size_t o = ((size_t)(n * CHN + c) * 64 + d) * 16 + s;
        H0[o] = H;
        H = Eb[o] + H * Pb[o];
    }
}

__global__ void k_scan3(const float* __restrict__ DT, const float* __restrict__ XDBL,
                        const _Float16* __restrict__ XCS, const float* __restrict__ Aneg,
                        const float* __restrict__ H0, float* __restrict__ YS) {
    int unit = blockIdx.x * 4 + (threadIdx.x >> 6);
    int n = unit >> 4, ch = unit & 15, d = threadIdx.x & 63;
    float Ad[16];
    #pragma unroll
    for (int s = 0; s < 16; s++) Ad[s] = Aneg[d * 16 + s];
    float h[16];
    size_t base = ((size_t)unit * 64 + d) * 16;
    #pragma unroll
    for (int s = 0; s < 16; s++) h[s] = H0[base + s];
    size_t row0 = (size_t)n * LL + ch * CLEN;
    for (int l = 0; l < CLEN; l++) {
        size_t row = row0 + l;
        size_t prow = row + 8 < (size_t)ROWS ? row + 8 : row;
        __builtin_prefetch(XDBL + prow * 48, 0, 1);
        __builtin_prefetch(DT + prow * 64 + d, 0, 1);
        float dt = DT[row * 64 + d];
        float dx = dt * (float)XCS[row * 64 + d];
        F16U UB, UC; const float4* pq = (const float4*)(XDBL + row * 48);
        UB.q[0] = pq[0]; UB.q[1] = pq[1]; UB.q[2] = pq[2]; UB.q[3] = pq[3];
        UC.q[0] = pq[4]; UC.q[1] = pq[5]; UC.q[2] = pq[6]; UC.q[3] = pq[7];
        float y = 0.f;
        #pragma unroll
        for (int s = 0; s < 16; s++) {
            float a = __expf(dt * Ad[s]);
            h[s] = h[s] * a + dx * UB.f[s];
            y += h[s] * UC.f[s];
        }
        YS[row * 64 + d] = y;
    }
}

// ---------------- K7: y = (ys + xcs*D) * silu(z) ----------------
__global__ void k_gate(const float* __restrict__ YS, const _Float16* __restrict__ XCS,
                       const float* __restrict__ XZ, const float* Dp,
                       _Float16* __restrict__ YG) {
    size_t i = (size_t)blockIdx.x * 256 + threadIdx.x;
    int d = (int)(i & 63); size_t row = i >> 6;
    float y = YS[i] + (float)XCS[i] * Dp[d];
    float z = XZ[row * 128 + 64 + d];
    y *= z * (1.f / (1.f + __expf(-z)));
    YG[i] = (_Float16)y;
}

// ---------------- K8: out_proj GEMM (64 -> 32) + skip; 4 WMMA/wave ----------------
__global__ void k_outproj(const _Float16* __restrict__ YG, const _Float16* __restrict__ W,
                          const float* __restrict__ XMF, const float* ssp,
                          float* __restrict__ YM) {
    int w = threadIdx.x >> 5, lane = threadIdx.x & 31;
    int mrow = (blockIdx.x * 8 + w) * 16;
    v16h a0 = ldfrag(YG, mrow, 64, 0);
    v16h a1 = ldfrag(YG, mrow, 64, 32);
    float ss = ssp[0];
    int mb = mrow + ((lane >> 4) << 3);
    int cl = lane & 15;
    #pragma unroll
    for (int nt = 0; nt < 2; nt++) {
        v8f c = {};
        c = wmma32(a0, ldfrag(W, nt * 16, 64, 0), c);
        c = wmma32(a1, ldfrag(W, nt * 16, 64, 32), c);
        int col = nt * 16 + cl;
        #pragma unroll
        for (int r = 0; r < 8; r++) {
            size_t o = (size_t)(mb + r) * 32 + col;
            YM[o] = c[r] + ss * XMF[o];
        }
    }
}

// ---------------- K9: regroup to (B,L,256) + LayerNorm ----------------
__global__ void k_ln2(const float* __restrict__ ym, const float* nw, const float* nb,
                      _Float16* __restrict__ yn) {
    __shared__ float sm[18];
    int blk = blockIdx.x; int b = blk >> 12; int l = blk & 4095; int c = threadIdx.x;
    float v = ym[((size_t)(b * 8 + (c >> 5)) * 4096 + l) * 32 + (c & 31)];
    float mu, rs; block_stats(v, mu, rs, sm);
    float o = (v - mu) * rs * nw[c] + nb[c];
    yn[(size_t)(b * 4096 + l) * 256 + c] = (_Float16)o;
}

// ---------------- K10: final proj (256 -> 256) + bias, NCHW store; 16 WMMA/wave ------
__global__ void k_proj(const _Float16* __restrict__ YN, const _Float16* __restrict__ W,
                       const float* __restrict__ pb, float* __restrict__ out) {
    int w = threadIdx.x >> 5, lane = threadIdx.x & 31;
    int mrow = blockIdx.x * 16;
    int nt0 = w, nt1 = w + 8;
    v8f c0 = {}, c1 = {};
    #pragma unroll
    for (int k = 0; k < 256; k += 32) {
        v16h a = ldfrag(YN, mrow, 256, k);
        v16h b0 = ldfrag(W, nt0 * 16, 256, k);
        c0 = wmma32(a, b0, c0);
        v16h b1 = ldfrag(W, nt1 * 16, 256, k);
        c1 = wmma32(a, b1, c1);
    }
    int mb = mrow + ((lane >> 4) << 3);
    int col0 = nt0 * 16 + (lane & 15), col1 = nt1 * 16 + (lane & 15);
    #pragma unroll
    for (int r = 0; r < 8; r++) {
        int rg = mb + r; int b = rg >> 12, l = rg & 4095;
        out[((size_t)(b * 256 + col0)) * 4096 + l] = c0[r] + pb[col0];
        out[((size_t)(b * 256 + col1)) * 4096 + l] = c1[r] + pb[col1];
    }
}

extern "C" void kernel_launch(void* const* d_in, const int* in_sizes, int n_in,
                              void* d_out, int out_size, void* d_ws, size_t ws_size,
                              hipStream_t stream) {
    (void)in_sizes; (void)n_in; (void)out_size; (void)ws_size;
    const float* x     = (const float*)d_in[0];
    const float* nw    = (const float*)d_in[1];
    const float* nb    = (const float*)d_in[2];
    const float* ss    = (const float*)d_in[3];
    const float* inw   = (const float*)d_in[4];
    const float* convw = (const float*)d_in[5];
    const float* convb = (const float*)d_in[6];
    const float* xpw   = (const float*)d_in[7];
    const float* dtw   = (const float*)d_in[8];
    const float* dtb   = (const float*)d_in[9];
    const float* alog  = (const float*)d_in[10];
    const float* Dp    = (const float*)d_in[11];
    const float* outw  = (const float*)d_in[12];
    const float* pw    = (const float*)d_in[13];
    const float* pb    = (const float*)d_in[14];
    float* out = (float*)d_out;

    char* ws = (char*)d_ws;
    float*    XMF  = (float*)   (ws + OFF_XM_F32);
    _Float16* XMH  = (_Float16*)(ws + OFF_XM_H);
    float*    XZ   = (float*)   (ws + OFF_XZ);
    _Float16* XCS  = (_Float16*)(ws + OFF_XCS_H);
    float*    XDBL = (float*)   (ws + OFF_XDBL);
    float*    DT   = (float*)   (ws + OFF_DT);
    float*    YS   = (float*)   (ws + OFF_YS);
    _Float16* YG   = (_Float16*)(ws + OFF_YG_H);
    float*    YM   = (float*)   (ws + OFF_YM);
    _Float16* YN   = (_Float16*)(ws + OFF_YN_H);
    float*    EB   = (float*)   (ws + OFF_EB);
    float*    PB   = (float*)   (ws + OFF_PB);
    float*    H0   = (float*)   (ws + OFF_H0);
    _Float16* WIN  = (_Float16*)(ws + OFF_WIN);
    _Float16* WXP  = (_Float16*)(ws + OFF_WXP);
    _Float16* WOUT = (_Float16*)(ws + OFF_WOUT);
    _Float16* WPJ  = (_Float16*)(ws + OFF_WPJ);
    float*    ANEG = (float*)   (ws + OFF_ANEG);

    k_prep<<<296, 256, 0, stream>>>(inw, xpw, outw, pw, alog, WIN, WXP, WOUT, WPJ, ANEG);
    k_ln1<<<16384, 256, 0, stream>>>(x, nw, nb, XMF, XMH);
    k_inproj<<<ROWS / 128, 256, 0, stream>>>(XMH, WIN, XZ);
    k_conv<<<(ROWS * 64) / 256, 256, 0, stream>>>(XZ, convw, convb, XCS);
    k_xproj<<<ROWS / 128, 256, 0, stream>>>(XCS, WXP, XDBL);
    k_dt<<<(ROWS * 64) / 256, 256, 0, stream>>>(XDBL, dtw, dtb, DT);
    k_scan1<<<NSEQ * CHN / 4, 256, 0, stream>>>(DT, XDBL, XCS, ANEG, EB, PB);
    k_scan2<<<(NSEQ * 64 * 16 + 255) / 256, 256, 0, stream>>>(EB, PB, H0);
    k_scan3<<<NSEQ * CHN / 4, 256, 0, stream>>>(DT, XDBL, XCS, ANEG, H0, YS);
    k_gate<<<(ROWS * 64) / 256, 256, 0, stream>>>(YS, XCS, XZ, Dp, YG);
    k_outproj<<<ROWS / 128, 256, 0, stream>>>(YG, WOUT, XMF, ss, YM);
    k_ln2<<<16384, 256, 0, stream>>>(YM, nw, nb, YN);
    k_proj<<<16384 / 16, 256, 0, stream>>>(YN, WPJ, pb, out);
}
